// RegressionLoss_27204322853526
// MI455X (gfx1250) — matrix operand order repaired and checked by reading, not compiled, not run on hardware
//
#include <hip/hip_runtime.h>
#include <hip/hip_bf16.h>

typedef _Float16 v16h __attribute__((ext_vector_type(16)));
typedef _Float16 h8   __attribute__((ext_vector_type(8)));
typedef float    v8f  __attribute__((ext_vector_type(8)));

#define NB   64      // total embeddings
#define DIM  512     // embedding dim
#define MB   32      // half batch
#define NP   256     // points per shape

__device__ __forceinline__ float wave_max32(float v) {
#pragma unroll
  for (int o = 16; o > 0; o >>= 1) v = fmaxf(v, __shfl_xor(v, o, 32));
  return v;
}
__device__ __forceinline__ float wave_sum32(float v) {
#pragma unroll
  for (int o = 16; o > 0; o >>= 1) v += __shfl_xor(v, o, 32);
  return v;
}

// ---------------------------------------------------------------------------
// 1) Row-normalize embeddings (64 x 512) and store an f16 copy for WMMA.
// ---------------------------------------------------------------------------
__global__ void normalize_to_half(const float* __restrict__ emb,
                                  _Float16* __restrict__ embH) {
  const int row = blockIdx.x;        // 0..63
  const int t   = threadIdx.x;       // 0..255
  __shared__ float red[256];
  const float x0 = emb[row * DIM + t];
  const float x1 = emb[row * DIM + 256 + t];
  red[t] = x0 * x0 + x1 * x1;
  __syncthreads();
#pragma unroll
  for (int s = 128; s > 0; s >>= 1) {
    if (t < s) red[t] += red[t + s];
    __syncthreads();
  }
  const float scale = rsqrtf(red[0]);
  embH[row * DIM + t]       = (_Float16)(x0 * scale);
  embH[row * DIM + 256 + t] = (_Float16)(x1 * scale);
}

// ---------------------------------------------------------------------------
// 2) inner_dot = sketch(32x512) @ shape(32x512)^T via v_wmma_f32_16x16x32_f16.
//    One block, 4 waves; wave w owns 16x16 tile (w>>1, w&1). K-loop of 16.
//    A order used here (contiguous K; AMD's table order differs): lane&15 = row M; lane>>4 selects K-half
//      (kb = 16*half); element e -> K = kb + e.
//    B order used here (same contiguous K as A): lane&15 = col N; lanes 0-15 hold K=0..15,
//      lanes 16-31 hold K=16..31 (element e -> K = 16*half + e).
// ---------------------------------------------------------------------------
__global__ void inner_dot_wmma(const _Float16* __restrict__ embH,
                               float* __restrict__ inner) {
  const int lane   = threadIdx.x & 31;
  const int wave   = threadIdx.x >> 5;        // 0..3
  const int rowbase = (wave >> 1) * 16;       // sketch tile base (M)
  const int colbase = (wave & 1) * 16;        // shape tile base (N)
  const int half16 = lane >> 4;               // 0 or 1
  const int m  = rowbase + (lane & 15);
  const int n  = colbase + (lane & 15);
  const int kbA = half16 * 16;
  const int kbB = half16 * 16;

  const _Float16* sk = embH;                  // rows 0..31  (sketch_feat)
  const _Float16* sh = embH + MB * DIM;       // rows 32..63 (shape_feat)

  v8f c = {};
#pragma unroll
  for (int k0 = 0; k0 < DIM; k0 += 32) {
    const h8 alo = *(const h8*)(sk + m * DIM + k0 + kbA);        // K = kb+0..7
    const h8 ahi = *(const h8*)(sk + m * DIM + k0 + kbA + 8);    // K = kb+8..15
    v16h a;
#pragma unroll
    for (int e = 0; e < 8; ++e) { a[e] = alo[e]; a[e + 8] = ahi[e]; }
    const v16h b = *(const v16h*)(sh + n * DIM + k0 + kbB);      // K = kbB+0..15
    c = __builtin_amdgcn_wmma_f32_16x16x32_f16(
        /*neg_a=*/false, a, /*neg_b=*/false, b,
        /*c_mod=*/(short)0, c, /*reuse_a=*/false, /*reuse_b=*/false);
  }
  // D layout: VGPR r -> M = rowbase + r + 8*half16, N = colbase + (lane&15)
#pragma unroll
  for (int r = 0; r < 8; ++r) {
    const int row = rowbase + r + half16 * 8;
    const int col = colbase + (lane & 15);
    inner[row * MB + col] = c[r];
  }
}

// ---------------------------------------------------------------------------
// 3) One-sided chamfer: t1[i,j] = mean_p min_q ||shape_i[p] - shape_j[q]||^2.
//    cd[i,j] = t1[i,j] + t1[j,i] by symmetry of d2 under (i,j,p,q)<->(j,i,q,p).
// ---------------------------------------------------------------------------
__global__ void chamfer_t1(const float* __restrict__ xyz,
                           float* __restrict__ t1) {
  const int j = blockIdx.x, i = blockIdx.y;   // pair (i, j)
  const int p = threadIdx.x;                  // 0..255
  const float* shapes = xyz + (size_t)MB * NP * 3;  // xyz[32:]
  __shared__ float qx[NP], qy[NP], qz[NP];
  __shared__ float red[NP];

  const float* qj = shapes + (size_t)j * NP * 3;
  qx[p] = qj[p * 3 + 0];
  qy[p] = qj[p * 3 + 1];
  qz[p] = qj[p * 3 + 2];
  __syncthreads();

  const float* pi = shapes + (size_t)i * NP * 3 + (size_t)p * 3;
  const float px = pi[0], py = pi[1], pz = pi[2];
  float mn = 3.4e38f;
#pragma unroll 4
  for (int q = 0; q < NP; ++q) {
    const float dx = px - qx[q], dy = py - qy[q], dz = pz - qz[q];
    mn = fminf(mn, dx * dx + dy * dy + dz * dz);
  }
  red[p] = mn;
  __syncthreads();
#pragma unroll
  for (int s = 128; s > 0; s >>= 1) {
    if (p < s) red[p] += red[p + s];
    __syncthreads();
  }
  if (p == 0) t1[i * MB + j] = red[0] * (1.0f / (float)NP);
}

// ---------------------------------------------------------------------------
// 4) Final: cd -> p = softmax(-cd^2/(2 sigma^2)); p_hat = softmax(inner);
//    reg = mean |p_hat - p|.  32 waves, wave r = row r, lane c = column c.
// ---------------------------------------------------------------------------
__global__ void finalize(const float* __restrict__ inner,
                         const float* __restrict__ t1,
                         float* __restrict__ out) {
  const int t = threadIdx.x;        // 0..1023
  const int r = t >> 5, c = t & 31; // wave32: one wave per row
  __shared__ float part[32];

  const float cd = t1[r * MB + c] + t1[c * MB + r];
  const float sigma = 1.0f * 0.997f / 3.0f;
  const float inv2s2 = 1.0f / (2.0f * sigma * sigma);
  const float x1 = -(cd * cd) * inv2s2;     // -real_dist
  const float x2 = inner[r * MB + c];       // inner_dot

  const float m1 = wave_max32(x1);
  const float e1 = __expf(x1 - m1);
  const float s1 = wave_sum32(e1);
  const float pv = e1 / s1;                 // p

  const float m2 = wave_max32(x2);
  const float e2 = __expf(x2 - m2);
  const float s2 = wave_sum32(e2);
  const float ph = e2 / s2;                 // p_hat

  const float wsum = wave_sum32(fabsf(ph - pv));
  if (c == 0) part[r] = wsum;
  __syncthreads();
  if (r == 0) {
    const float v = wave_sum32(part[c]);
    if (c == 0) out[0] = v * (1.0f / 1024.0f);
  }
}

// ---------------------------------------------------------------------------
extern "C" void kernel_launch(void* const* d_in, const int* in_sizes, int n_in,
                              void* d_out, int out_size, void* d_ws, size_t ws_size,
                              hipStream_t stream) {
  const float* emb = (const float*)d_in[0];   // (64, 512) f32
  const float* xyz = (const float*)d_in[1];   // (64, 256, 3) f32

  _Float16* embH = (_Float16*)d_ws;                                   // 64*512 f16
  float* inner = (float*)((char*)d_ws + (size_t)NB * DIM * sizeof(_Float16)); // 32*32
  float* t1    = inner + MB * MB;                                     // 32*32

  normalize_to_half<<<NB, 256, 0, stream>>>(emb, embH);
  inner_dot_wmma<<<1, 128, 0, stream>>>(embH, inner);
  chamfer_t1<<<dim3(MB, MB), NP, 0, stream>>>(xyz, t1);
  finalize<<<1, 1024, 0, stream>>>(inner, t1, (float*)d_out);
}
